// LSTM_only_1752346657197
// MI455X (gfx1250) — matrix operand order repaired and checked by reading, not compiled, not run on hardware
//
#include <hip/hip_runtime.h>

typedef __attribute__((ext_vector_type(16))) __bf16 v16bf;
typedef __attribute__((ext_vector_type(8)))  float  v8f;

#define BATCH 2048
#define HID   128
#define G4    512
#define TSTEPS 28
#define IDIM  28

__device__ __forceinline__ float fastrcp(float x) {
    return __builtin_amdgcn_rcpf(x);
}
__device__ __forceinline__ float sigf(float x) {
    return fastrcp(1.0f + __expf(-x));
}
__device__ __forceinline__ float tanhfast(float x) {
    x = fminf(fmaxf(x, -15.0f), 15.0f);
    float e = __expf(2.0f * x);
    return (e - 1.0f) * fastrcp(e + 1.0f);
}

__device__ __forceinline__ v16bf load_frag16(const __bf16* p0, const __bf16* p1) {
    union { v16bf v; uint4 u[2]; } t;
    t.u[0] = *reinterpret_cast<const uint4*>(p0);
    t.u[1] = *reinterpret_cast<const uint4*>(p1);
    return t.v;
}

// ---------------------------------------------------------------------------
// Prep: convert the four Whh (512x128 f32) to bf16 in workspace (L2-resident)
// ---------------------------------------------------------------------------
__global__ __launch_bounds__(256) void prep_whh_bf16(
    const float* __restrict__ a0, const float* __restrict__ a1,
    const float* __restrict__ a2, const float* __restrict__ a3,
    __bf16* __restrict__ out)
{
    int dir = blockIdx.y;
    const float* src = dir == 0 ? a0 : dir == 1 ? a1 : dir == 2 ? a2 : a3;
    int i = blockIdx.x * 256 + threadIdx.x;           // 0 .. 65535
    out[(size_t)dir * (G4 * HID) + i] = (__bf16)src[i];
}

// ---------------------------------------------------------------------------
// Fused 4-direction LSTM scan. grid = (B/16, 4 dirs), block = 256 (8 waves).
// Each workgroup owns 16 batch rows of one direction for the full 28 steps.
// All 28 input A-tiles staged in LDS up-front; h double-buffered in LDS so the
// scan needs only ONE barrier per step. Wave w computes gate columns
// {w*16..w*16+15} of i,f,g,o -> cell update fully in-register (c stays in the
// WMMA accumulator layout the whole scan).
// ---------------------------------------------------------------------------
__global__ __launch_bounds__(256) void lstm4_scan_kernel(
    const float* __restrict__ x,
    const float* __restrict__ wih0, const float* __restrict__ wih1,
    const float* __restrict__ wih2, const float* __restrict__ wih3,
    const float* __restrict__ bih0, const float* __restrict__ bih1,
    const float* __restrict__ bih2, const float* __restrict__ bih3,
    const float* __restrict__ bhh0, const float* __restrict__ bhh1,
    const float* __restrict__ bhh2, const float* __restrict__ bhh3,
    const float* __restrict__ h0lr, const float* __restrict__ c0lr,
    const float* __restrict__ h0ud, const float* __restrict__ c0ud,
    const __bf16* __restrict__ whhbf,   // [4][512][128] bf16
    float* __restrict__ featT)          // [512][2048] f32
{
    __shared__ __align__(16) __bf16 s_aall[TSTEPS][16][32]; // 28 KB: all A tiles
    __shared__ __align__(16) __bf16 s_h[2][16][HID];        // 8 KB: ping-pong h

    const int tid  = threadIdx.x;
    const int dir  = blockIdx.y;
    const int b0   = blockIdx.x * 16;
    const int lane = tid & 31;
    const int w    = tid >> 5;        // wave 0..7
    const int nn   = lane & 15;
    const int hi   = lane >> 4;

    const float* Wih = dir == 0 ? wih0 : dir == 1 ? wih1 : dir == 2 ? wih2 : wih3;
    const float* bih = dir == 0 ? bih0 : dir == 1 ? bih1 : dir == 2 ? bih2 : bih3;
    const float* bhh = dir == 0 ? bhh0 : dir == 1 ? bhh1 : dir == 2 ? bhh2 : bhh3;
    const float* h0  = (dir < 2 ? h0lr : h0ud) + (size_t)(dir & 1) * (BATCH * HID);
    const float* c0  = (dir < 2 ? c0lr : c0ud) + (size_t)(dir & 1) * (BATCH * HID);
    const __bf16* whh = whhbf + (size_t)dir * (G4 * HID);

    // ---- stage ALL 28 per-step A tiles (direction-aware x slicing) ----
    for (int idx = tid; idx < TSTEPS * 16 * 32; idx += 256) {
        int t   = idx >> 9;
        int rem = idx & 511;
        int m   = rem >> 5, kk = rem & 31;
        float v = 0.0f;
        if (kk < IDIM) {
            int tt = (dir & 1) ? (TSTEPS - 1 - t) : t;
            int p  = (dir < 2) ? (tt * IDIM + kk) : (kk * IDIM + tt);
            v = x[(size_t)(b0 + m) * 784 + p];
        }
        s_aall[t][m][kk] = (__bf16)v;
    }
    // ---- stage h0 into ping buffer 0 ----
    for (int idx = tid; idx < 16 * HID; idx += 256) {
        int m = idx >> 7, k = idx & 127;
        s_h[0][m][k] = (__bf16)h0[(size_t)(b0 + m) * HID + k];
    }

    // ---- per-wave bias (lane holds column n = gi*128 + w*16 + nn) ----
    float bias[4];
#pragma unroll
    for (int gi = 0; gi < 4; ++gi) {
        int n = gi * HID + w * 16 + nn;
        bias[gi] = bih[n] + bhh[n];
    }
    // ---- c state lives in WMMA C/D layout: VGPR r -> row (r + 8*hi) ----
    v8f c;
#pragma unroll
    for (int r = 0; r < 8; ++r)
        c[r] = c0[(size_t)(b0 + r + 8 * hi) * HID + (w * 16 + nn)];

    // ---- input-projection B fragments straight from global f32 (one-time) ----
    v16bf wf[4];
#pragma unroll
    for (int gi = 0; gi < 4; ++gi) {
        int n = gi * HID + w * 16 + nn;
#pragma unroll
        for (int e = 0; e < 16; ++e) {
            int k = hi * 8 + (e < 8 ? e : e + 8);
            wf[gi][e] = (__bf16)(k < IDIM ? Wih[n * IDIM + k] : 0.0f);
        }
    }

    float hout[8];
    const int abase = hi * 8;

    for (int t = 0; t < TSTEPS; ++t) {
        __syncthreads();   // h[cur] writes from prev step visible everywhere
        const int cur = t & 1;

        // ---- A fragments (ISA 16-bit A 16x32 layout) ----
        v16bf af = load_frag16(&s_aall[t][nn][abase], &s_aall[t][nn][abase + 16]);
        v16bf hf[4];
#pragma unroll
        for (int kf = 0; kf < 4; ++kf) {
            int base = kf * 32 + abase;
            hf[kf] = load_frag16(&s_h[cur][nn][base], &s_h[cur][nn][base + 16]);
        }

        // ---- gate GEMMs: 1 input-proj WMMA + 4 recurrent WMMAs per tile ----
        v8f acc[4];
#pragma unroll
        for (int gi = 0; gi < 4; ++gi) {
            int n = gi * HID + w * 16 + nn;
            v8f a;
#pragma unroll
            for (int r = 0; r < 8; ++r) a[r] = bias[gi];
            a = __builtin_amdgcn_wmma_f32_16x16x32_bf16(
                    false, af, false, wf[gi], (short)0, a, false, false);
#pragma unroll
            for (int kf = 0; kf < 4; ++kf) {
                const __bf16* wp = whh + (size_t)n * HID + kf * 32 + hi * 8;
                v16bf bfrag = load_frag16(wp, wp + 16);
                a = __builtin_amdgcn_wmma_f32_16x16x32_bf16(
                        false, hf[kf], false, bfrag, (short)0, a, false, false);
            }
            acc[gi] = a;
        }

        // ---- in-register cell/hidden update (i,f,g,o all in this wave) ----
#pragma unroll
        for (int r = 0; r < 8; ++r) {
            float iv = sigf(acc[0][r]);
            float fv = sigf(acc[1][r]);
            float gv = tanhfast(acc[2][r]);
            float ov = sigf(acc[3][r]);
            float cn = fv * c[r] + iv * gv;
            c[r] = cn;
            float hv = ov * tanhfast(cn);
            hout[r] = hv;
            s_h[1 - cur][r + 8 * hi][w * 16 + nn] = (__bf16)hv;
        }
    }

    // ---- final h -> transposed feature buffer featT[dir*128 + j][b] ----
    int row = dir * HID + w * 16 + nn;
#pragma unroll
    for (int r = 0; r < 8; ++r)
        featT[(size_t)row * BATCH + b0 + r + 8 * hi] = hout[r];
}

// ---------------------------------------------------------------------------
// fc-LSTM (T=512, HF=10) + log_softmax. 4 lanes cooperate per batch row:
// lane q owns gate-type q (10 gates), weights cached in VGPRs, gate exchange
// via wave shuffles. 8192 threads = 256 waves, zero LDS in the hot loop.
// ---------------------------------------------------------------------------
__global__ __launch_bounds__(256) void fc_lstm_kernel(
    const float* __restrict__ featT,   // [512][2048]
    const float* __restrict__ wihfc,   // [40]
    const float* __restrict__ whhfc,   // [40][10]
    const float* __restrict__ bihfc, const float* __restrict__ bhhfc,
    const float* __restrict__ h0, const float* __restrict__ c0,
    float* __restrict__ out)           // [2048][10]
{
    int gtid = blockIdx.x * 256 + threadIdx.x;   // 0..8191
    int b = gtid >> 2;
    int q = gtid & 3;
    int lane = threadIdx.x & 31;
    int laneBase = lane & ~3;

    float w[10][10], wx[10], bs[10];
#pragma unroll
    for (int j = 0; j < 10; ++j) {
        int n = q * 10 + j;
        wx[j] = wihfc[n];
        bs[j] = bihfc[n] + bhhfc[n];
#pragma unroll
        for (int k = 0; k < 10; ++k) w[j][k] = whhfc[n * 10 + k];
    }
    float h[10], c[10];
#pragma unroll
    for (int j = 0; j < 10; ++j) {
        h[j] = h0[b * 10 + j];
        c[j] = c0[b * 10 + j];
    }

    for (int t = 0; t < 512; ++t) {
        float xv = featT[(size_t)t * BATCH + b];
        float g[10];
#pragma unroll
        for (int j = 0; j < 10; ++j) {
            float a = bs[j] + wx[j] * xv;
#pragma unroll
            for (int k = 0; k < 10; ++k) a += w[j][k] * h[k];
            g[j] = a;
        }
#pragma unroll
        for (int j = 0; j < 10; ++j) {
            float iv = __shfl(g[j], laneBase + 0, 32);
            float fv = __shfl(g[j], laneBase + 1, 32);
            float gv = __shfl(g[j], laneBase + 2, 32);
            float ov = __shfl(g[j], laneBase + 3, 32);
            float cn = sigf(fv) * c[j] + sigf(iv) * tanhfast(gv);
            c[j] = cn;
            h[j] = sigf(ov) * tanhfast(cn);
        }
    }

    if (q == 0) {
        float m = h[0];
#pragma unroll
        for (int j = 1; j < 10; ++j) m = fmaxf(m, h[j]);
        float s = 0.0f;
#pragma unroll
        for (int j = 0; j < 10; ++j) s += __expf(h[j] - m);
        float ls = __logf(s);
#pragma unroll
        for (int j = 0; j < 10; ++j) out[b * 10 + j] = h[j] - m - ls;
    }
}

// ---------------------------------------------------------------------------
extern "C" void kernel_launch(void* const* d_in, const int* in_sizes, int n_in,
                              void* d_out, int out_size, void* d_ws, size_t ws_size,
                              hipStream_t stream) {
    (void)in_sizes; (void)n_in; (void)out_size; (void)ws_size;

    const float* x        = (const float*)d_in[0];
    const float* wih[4]   = { (const float*)d_in[1],  (const float*)d_in[5],
                              (const float*)d_in[9],  (const float*)d_in[13] };
    const float* whh[4]   = { (const float*)d_in[2],  (const float*)d_in[6],
                              (const float*)d_in[10], (const float*)d_in[14] };
    const float* bihv[4]  = { (const float*)d_in[3],  (const float*)d_in[7],
                              (const float*)d_in[11], (const float*)d_in[15] };
    const float* bhhv[4]  = { (const float*)d_in[4],  (const float*)d_in[8],
                              (const float*)d_in[12], (const float*)d_in[16] };
    const float* wih_fc = (const float*)d_in[17];
    const float* whh_fc = (const float*)d_in[18];
    const float* bih_fc = (const float*)d_in[19];
    const float* bhh_fc = (const float*)d_in[20];
    const float* h0_lr  = (const float*)d_in[21];
    const float* c0_lr  = (const float*)d_in[22];
    const float* h0_ud  = (const float*)d_in[23];
    const float* c0_ud  = (const float*)d_in[24];
    const float* h0_fc  = (const float*)d_in[25];
    const float* c0_fc  = (const float*)d_in[26];

    char*   ws    = (char*)d_ws;
    float*  featT = (float*)ws;                                   // 4 MB
    __bf16* whhbf = (__bf16*)(ws + (size_t)G4 * BATCH * 4);       // 512 KB

    prep_whh_bf16<<<dim3(256, 4), 256, 0, stream>>>(
        whh[0], whh[1], whh[2], whh[3], whhbf);

    lstm4_scan_kernel<<<dim3(BATCH / 16, 4), 256, 0, stream>>>(
        x,
        wih[0], wih[1], wih[2], wih[3],
        bihv[0], bihv[1], bihv[2], bihv[3],
        bhhv[0], bhhv[1], bhhv[2], bhhv[3],
        h0_lr, c0_lr, h0_ud, c0_ud,
        whhbf, featT);

    fc_lstm_kernel<<<dim3(8192 / 256), 256, 0, stream>>>(
        featT, wih_fc, whh_fc, bih_fc, bhh_fc, h0_fc, c0_fc, (float*)d_out);
}
